// MultiHeadAttention_86199993630904
// MI455X (gfx1250) — matrix operand order repaired and checked by reading, not compiled, not run on hardware
//
#include <hip/hip_runtime.h>
#include <hip/hip_bf16.h>
#include <math.h>
#include <stdint.h>

// Problem constants (match reference)
#define B_  2
#define L_  2048
#define D_  1024
#define H_  16
#define HD_ 64

typedef __attribute__((ext_vector_type(16))) __bf16 v16bf;
typedef __attribute__((ext_vector_type(8)))  float  v8f;

// Tensor-DMA descriptor vector types
typedef __attribute__((ext_vector_type(4))) unsigned int u32x4;
typedef __attribute__((ext_vector_type(4))) int          i32x4;
typedef __attribute__((ext_vector_type(8))) int          i32x8;

#if defined(__has_builtin)
#if __has_builtin(__builtin_amdgcn_tensor_load_to_lds) && __has_builtin(__builtin_amdgcn_s_wait_tensorcnt)
#define USE_TDM 1
#endif
#endif
#ifndef USE_TDM
#define USE_TDM 0
#endif

// A-operand (16xK, 16-bit) slot->k mapping for one K=32 chunk.
// lane 0-15: K = {0..7, 16..23}; lane 16-31: K = {8..15, 24..31}
__device__ __forceinline__ int kmapA(int j, int kbase) {
    return ((j >> 3) << 4) + kbase + (j & 7);
}
// B-operand (Kx16, 16-bit) slot->k mapping for one K=32 chunk.
// lane 0-15: K = {0..7, 16..23}; lane 16-31: K = {8..15, 24..31}
__device__ __forceinline__ int kmapB(int j, int hi) {
    return ((j >> 3) << 4) + (hi << 3) + (j & 7);
}

#if USE_TDM
// TDM copy of a 2D bf16 tile (tileW elems x tileH rows, row stride strideElems)
// from global to LDS. D# per CDNA5 ISA 8.3/8.4. tileW*2 bytes must be 4B-multiple.
__device__ __forceinline__ void tdm_copy_2d_bf16(unsigned ldsOff, const __bf16* g,
                                                 int tileW, int tileH, int strideElems) {
    unsigned long long ga = (unsigned long long)(uintptr_t)g;
    u32x4 g0;
    g0[0] = 1u;                                             // count=1 (valid user D#)
    g0[1] = ldsOff;                                         // lds_addr (bytes)
    g0[2] = (unsigned)(ga & 0xffffffffu);                   // global_addr[31:0]
    g0[3] = (unsigned)((ga >> 32) & 0x01ffffffu)            // global_addr[56:32]
            | (2u << 30);                                   // type = 2 ("image")
    i32x8 g1;
    g1[0] = 1 << 16;                                        // data_size=1 (2B); wg_mask=0
    g1[1] = (strideElems & 0xffff) << 16;                   // tensor_dim0[15:0]
    g1[2] = ((strideElems >> 16) & 0xffff)                  // tensor_dim0[31:16]
            | ((tileH & 0xffff) << 16);                     // tensor_dim1[15:0] = tileH
    g1[3] = (tileW & 0xffff) << 16;                         // tensor_dim1[31:16]=0 | tile_dim0
    g1[4] = tileH & 0xffff;                                 // tile_dim1 | tile_dim2=0
    g1[5] = strideElems;                                    // tensor_dim0_stride[31:0]
    g1[6] = (strideElems & 0xffff) << 16;                   // stride0[47:32]=0 | stride1[15:0]
    g1[7] = 0;
    i32x4 z4 = {0, 0, 0, 0};
#if __clang_major__ >= 23
    i32x8 z8 = {0, 0, 0, 0, 0, 0, 0, 0};
    __builtin_amdgcn_tensor_load_to_lds(g0, g1, z4, z4, z8, 0);
#else
    __builtin_amdgcn_tensor_load_to_lds(g0, g1, z4, z4, 0);
#endif
}
#endif

// ---------------------------------------------------------------------------
// Kernel 0: Wo f32 -> bf16
// ---------------------------------------------------------------------------
__global__ __launch_bounds__(256) void cvt_wo_kernel(const float* __restrict__ Wo,
                                                     __bf16* __restrict__ Wob) {
    int i = blockIdx.x * 256 + threadIdx.x;
    if (i < D_ * D_) Wob[i] = (__bf16)Wo[i];
}

// ---------------------------------------------------------------------------
// Kernel 1: fused per-head QKV projection.
// grid = (L/128, B*H), block = 256 (8 waves). q = x_head @ Wq^T etc.
// ---------------------------------------------------------------------------
__global__ __launch_bounds__(256) void qkv_proj_kernel(
    const float* __restrict__ x,
    const float* __restrict__ Wq, const float* __restrict__ Wk, const float* __restrict__ Wv,
    __bf16* __restrict__ Qw, __bf16* __restrict__ Kw, __bf16* __restrict__ Vw)
{
    __shared__ __bf16 xs[128 * 64];
    __shared__ __bf16 wsh[3][64 * 64];

    const int bh    = blockIdx.y;
    const int b     = bh / H_;
    const int h     = bh % H_;
    const int lbase = blockIdx.x * 128;
    const int tid   = threadIdx.x;

    // Stage x head-slice tile (128 rows x 64 cols) as bf16 (f32->bf16: no TDM here)
    for (int i = tid; i < 128 * 64; i += 256) {
        int row = i >> 6, col = i & 63;
        xs[i] = (__bf16)x[((size_t)b * L_ + (lbase + row)) * D_ + h * HD_ + col];
    }
    for (int i = tid; i < 64 * 64; i += 256) {
        wsh[0][i] = (__bf16)Wq[i];
        wsh[1][i] = (__bf16)Wk[i];
        wsh[2][i] = (__bf16)Wv[i];
    }
    __syncthreads();

    const int wave  = tid >> 5;
    const int lane  = tid & 31;
    const int nloc  = lane & 15;
    const int hi    = lane >> 4;
    const int kbase = hi << 3;

    // 3 matrices x (8 row-tiles x 4 col-tiles) = 96 16x16 output tiles
    for (int t = wave; t < 96; t += 8) {
        const int mat  = t >> 5;
        const int tt   = t & 31;
        const int mrow = (tt >> 2) << 4;   // 0..112
        const int ncol = (tt & 3) << 4;    // 0..48

        v8f c;
        #pragma unroll
        for (int r = 0; r < 8; ++r) c[r] = 0.0f;

        #pragma unroll
        for (int ch = 0; ch < 2; ++ch) {
            v16bf a, bfrag;
            #pragma unroll
            for (int j = 0; j < 16; ++j) {
                a[j]     = xs[(mrow + nloc) * 64 + ch * 32 + kmapA(j, kbase)];
                // B[k=d][n=e] = W[e][d]  (y = x @ W^T)
                bfrag[j] = wsh[mat][(ncol + nloc) * 64 + ch * 32 + kmapB(j, hi)];
            }
            c = __builtin_amdgcn_wmma_f32_16x16x32_bf16(false, a, false, bfrag,
                                                        (short)0, c, false, false);
        }

        __bf16* dst = (mat == 0) ? Qw : (mat == 1) ? Kw : Vw;
        #pragma unroll
        for (int r = 0; r < 8; ++r) {
            int row = lbase + mrow + r + (hi << 3);
            dst[((size_t)bh * L_ + row) * HD_ + ncol + nloc] = (__bf16)c[r];
        }
    }
}

// ---------------------------------------------------------------------------
// Kernel 2: flash attention with the reference's quirky masking.
// grid = B*H*(L/64), block = 128 (4 waves); each wave owns 16 queries.
// K/V tiles double-buffered in LDS; Tensor Data Mover pipelines tile i+1
// while the block computes on tile i (in-order TENSORcnt: wait <=2).
// ---------------------------------------------------------------------------
__global__ __launch_bounds__(128) void flash_attn_kernel(
    const __bf16* __restrict__ Qw, const __bf16* __restrict__ Kw, const __bf16* __restrict__ Vw,
    const int* __restrict__ padding_mask, const int* __restrict__ future_mask,
    __bf16* __restrict__ Cw)
{
    __shared__ __bf16 kt[2][32 * 64];    // K tile double buffer (32 keys x 64 dims)
    __shared__ __bf16 vt[2][32 * 64];    // V tile double buffer
    __shared__ __bf16 plds[4][16 * 32];  // per-wave P tile for C->A relayout

    const int tid   = threadIdx.x;
    const int wave  = tid >> 5;
    const int lane  = tid & 31;
    const int nloc  = lane & 15;
    const int hi    = lane >> 4;
    const int kbase = hi << 3;

    const int nQblk = L_ / 64;               // 32
    const int bh    = blockIdx.x / nQblk;
    const int qblk  = blockIdx.x % nQblk;
    const int b     = bh / H_;
    const int h     = bh % H_;
    const int qbase = qblk * 64 + wave * 16;

    const __bf16* Qp    = Qw + (size_t)bh * L_ * HD_;
    const __bf16* Kp    = Kw + (size_t)bh * L_ * HD_;
    const __bf16* Vp    = Vw + (size_t)bh * L_ * HD_;
    const int*    pmask = padding_mask + (size_t)b * L_;

    // Q A-fragments (16 x 64 -> two K=32 chunks), held in registers all loop
    v16bf aq[2];
    const int qrow = qbase + nloc;
    #pragma unroll
    for (int ch = 0; ch < 2; ++ch)
        #pragma unroll
        for (int j = 0; j < 16; ++j)
            aq[ch][j] = Qp[(size_t)qrow * HD_ + ch * 32 + kmapA(j, kbase)];

    float mrow[8], lrow[8];
    v8f o[4];
    #pragma unroll
    for (int r = 0; r < 8; ++r) { mrow[r] = -__builtin_inff(); lrow[r] = 0.0f; }
    #pragma unroll
    for (int t = 0; t < 4; ++t)
        #pragma unroll
        for (int r = 0; r < 8; ++r) o[t][r] = 0.0f;

    const int nIter = L_ / 32;   // 64

#if USE_TDM
    const unsigned ktOff0 = (unsigned)(uintptr_t)&kt[0][0];
    const unsigned ktOff1 = (unsigned)(uintptr_t)&kt[1][0];
    const unsigned vtOff0 = (unsigned)(uintptr_t)&vt[0][0];
    const unsigned vtOff1 = (unsigned)(uintptr_t)&vt[1][0];
    if (wave == 0) {    // prologue: DMA tile 0 into buffer 0
        tdm_copy_2d_bf16(ktOff0, Kp, 32 * HD_, 1, 32 * HD_);
        tdm_copy_2d_bf16(vtOff0, Vp, 32 * HD_, 1, 32 * HD_);
    }
#endif

    for (int it = 0; it < nIter; ++it) {
        const int kb  = it * 32;
        const int buf = it & 1;

        __syncthreads();  // all waves done reading buffer buf^1 (tile it-1)
#if USE_TDM
        if (wave == 0) {
            if (it + 1 < nIter) {   // DMA tile it+1 into the other buffer
                const __bf16* nk = Kp + (size_t)(kb + 32) * HD_;
                const __bf16* nv = Vp + (size_t)(kb + 32) * HD_;
                tdm_copy_2d_bf16(buf ? ktOff0 : ktOff1, nk, 32 * HD_, 1, 32 * HD_);
                tdm_copy_2d_bf16(buf ? vtOff0 : vtOff1, nv, 32 * HD_, 1, 32 * HD_);
                __builtin_amdgcn_s_wait_tensorcnt(2);   // in-order: tile it landed
            } else {
                __builtin_amdgcn_s_wait_tensorcnt(0);
            }
        }
#else
        {   // fallback: cooperative, coalesced dword staging (contiguous 4KB each)
            const unsigned int* ks = (const unsigned int*)(Kp + (size_t)kb * HD_);
            const unsigned int* vs = (const unsigned int*)(Vp + (size_t)kb * HD_);
            unsigned int* kd = (unsigned int*)kt[buf];
            unsigned int* vd = (unsigned int*)vt[buf];
            for (int i = tid; i < 32 * 64 / 2; i += 128) { kd[i] = ks[i]; vd[i] = vs[i]; }
        }
#endif
        __syncthreads();  // tile it visible to all waves

        const __bf16* ktb = kt[buf];
        const __bf16* vtb = vt[buf];

        // S = Q K^T : two 16x16 tiles (key halves), each 2 WMMAs over K-dim 64
        v8f s[2];
        #pragma unroll
        for (int g = 0; g < 2; ++g) {
            v8f c;
            #pragma unroll
            for (int r = 0; r < 8; ++r) c[r] = 0.0f;
            const int keyl = g * 16 + nloc;         // B lane column = key row
            #pragma unroll
            for (int ch = 0; ch < 2; ++ch) {
                v16bf bk;
                #pragma unroll
                for (int j = 0; j < 16; ++j)
                    bk[j] = ktb[keyl * 64 + ch * 32 + kmapB(j, hi)];
                c = __builtin_amdgcn_wmma_f32_16x16x32_bf16(false, aq[ch], false, bk,
                                                            (short)0, c, false, false);
            }
            s[g] = c;
        }

        // Faithful masking BEFORE scaling: (future+padding)>1 -> -1e20, then /8
        #pragma unroll
        for (int g = 0; g < 2; ++g) {
            const int key = kb + g * 16 + nloc;
            const int pm  = pmask[key];
            #pragma unroll
            for (int r = 0; r < 8; ++r) {
                const int q = qbase + r + (hi << 3);
                float sv = s[g][r];
                if (future_mask[(size_t)q * L_ + key] + pm > 1) sv = -1.0e20f;
                s[g][r] = sv * 0.125f;
            }
        }

        // Online softmax stats per row (row m = r + 8*hi lives across 16 lanes)
        float alpha[8];
        #pragma unroll
        for (int r = 0; r < 8; ++r) {
            float mx = fmaxf(s[0][r], s[1][r]);
            #pragma unroll
            for (int off = 1; off < 16; off <<= 1)
                mx = fmaxf(mx, __shfl_xor(mx, off, 32));
            const float mnew = fmaxf(mrow[r], mx);
            const float a    = __expf(mrow[r] - mnew);
            const float p0   = __expf(s[0][r] - mnew);
            const float p1   = __expf(s[1][r] - mnew);
            s[0][r] = p0; s[1][r] = p1;
            float rs = p0 + p1;
            #pragma unroll
            for (int off = 1; off < 16; off <<= 1)
                rs += __shfl_xor(rs, off, 32);
            lrow[r]  = lrow[r] * a + rs;
            mrow[r]  = mnew;
            alpha[r] = a;
        }
        #pragma unroll
        for (int t = 0; t < 4; ++t)
            #pragma unroll
            for (int r = 0; r < 8; ++r) o[t][r] *= alpha[r];

        // P: C-layout -> row-major LDS (bf16) -> reload in A-layout.
        // plds[wave] is wave-private; same-wave DS ops are serviced in order,
        // so a compiler ordering barrier suffices (no workgroup s_barrier).
        __bf16* pl = plds[wave];
        #pragma unroll
        for (int r = 0; r < 8; ++r) {
            const int m = r + (hi << 3);
            pl[m * 32 + nloc]      = (__bf16)s[0][r];
            pl[m * 32 + 16 + nloc] = (__bf16)s[1][r];
        }
        asm volatile("" ::: "memory");
        v16bf pa;
        #pragma unroll
        for (int j = 0; j < 16; ++j)
            pa[j] = pl[nloc * 32 + kmapA(j, kbase)];

        // O += P @ V   (four 16x16 column tiles of the 16x64 accumulator)
        #pragma unroll
        for (int t = 0; t < 4; ++t) {
            v16bf bv;
            #pragma unroll
            for (int j = 0; j < 16; ++j)
                bv[j] = vtb[(size_t)kmapB(j, hi) * 64 + t * 16 + nloc];
            o[t] = __builtin_amdgcn_wmma_f32_16x16x32_bf16(false, pa, false, bv,
                                                           (short)0, o[t], false, false);
        }
    }

    // Normalize and store concat (B, L, H*HD) as bf16 for the output projection
    #pragma unroll
    for (int r = 0; r < 8; ++r) {
        const float inv = 1.0f / lrow[r];
        const int q = qbase + r + (hi << 3);
        #pragma unroll
        for (int t = 0; t < 4; ++t)
            Cw[((size_t)b * L_ + q) * D_ + h * HD_ + t * 16 + nloc] = (__bf16)(o[t][r] * inv);
    }
}

// ---------------------------------------------------------------------------
// Kernel 3: output projection  out = concat @ Wo^T + bo
// grid = (B*L/64, D/64), block = 128 (4 waves); 64x64 tile per block.
// A/B K=32 slabs are true 2D strided tiles -> 2D TDM descriptors, double
// buffered and pipelined on TENSORcnt like the attention kernel.
// ---------------------------------------------------------------------------
__global__ __launch_bounds__(128) void out_proj_kernel(
    const __bf16* __restrict__ Cw, const __bf16* __restrict__ Wob,
    const float* __restrict__ bo, float* __restrict__ out)
{
    __shared__ __bf16 as[2][64 * 32];   // activations: 64 rows x 32 k
    __shared__ __bf16 bs[2][64 * 32];   // Wo rows (= output cols): 64 x 32 k

    const int tid   = threadIdx.x;
    const int wave  = tid >> 5;
    const int lane  = tid & 31;
    const int nloc  = lane & 15;
    const int hi    = lane >> 4;
    const int kbase = hi << 3;

    const int rowbase = blockIdx.x * 64;
    const int colbase = blockIdx.y * 64;

    const __bf16* Ap = Cw + (size_t)rowbase * D_;
    const __bf16* Bp = Wob + (size_t)colbase * D_;

    v8f acc[4];
    #pragma unroll
    for (int t = 0; t < 4; ++t)
        #pragma unroll
        for (int r = 0; r < 8; ++r) acc[t][r] = 0.0f;

    const int nIter = D_ / 32;   // 32

#if USE_TDM
    const unsigned asOff0 = (unsigned)(uintptr_t)&as[0][0];
    const unsigned asOff1 = (unsigned)(uintptr_t)&as[1][0];
    const unsigned bsOff0 = (unsigned)(uintptr_t)&bs[0][0];
    const unsigned bsOff1 = (unsigned)(uintptr_t)&bs[1][0];
    if (wave == 0) {    // prologue: DMA K-slab 0 into buffer 0
        tdm_copy_2d_bf16(asOff0, Ap, 32, 64, D_);
        tdm_copy_2d_bf16(bsOff0, Bp, 32, 64, D_);
    }
#endif

    for (int it = 0; it < nIter; ++it) {
        const int k0  = it * 32;
        const int buf = it & 1;

        __syncthreads();
#if USE_TDM
        if (wave == 0) {
            if (it + 1 < nIter) {
                tdm_copy_2d_bf16(buf ? asOff0 : asOff1, Ap + k0 + 32, 32, 64, D_);
                tdm_copy_2d_bf16(buf ? bsOff0 : bsOff1, Bp + k0 + 32, 32, 64, D_);
                __builtin_amdgcn_s_wait_tensorcnt(2);
            } else {
                __builtin_amdgcn_s_wait_tensorcnt(0);
            }
        }
#else
        for (int i = tid; i < 64 * 32; i += 128) {
            const int r = i >> 5, c = i & 31;
            as[buf][i] = Ap[(size_t)r * D_ + k0 + c];
            bs[buf][i] = Bp[(size_t)r * D_ + k0 + c];   // y=x@Wo^T -> B col n = Wo row n
        }
#endif
        __syncthreads();

        v16bf a;
        #pragma unroll
        for (int j = 0; j < 16; ++j)
            a[j] = as[buf][(wave * 16 + nloc) * 32 + kmapA(j, kbase)];
        #pragma unroll
        for (int t = 0; t < 4; ++t) {
            v16bf bfrag;
            #pragma unroll
            for (int j = 0; j < 16; ++j)
                bfrag[j] = bs[buf][(t * 16 + nloc) * 32 + kmapB(j, hi)];
            acc[t] = __builtin_amdgcn_wmma_f32_16x16x32_bf16(false, a, false, bfrag,
                                                             (short)0, acc[t], false, false);
        }
    }

    #pragma unroll
    for (int r = 0; r < 8; ++r) {
        const int row = rowbase + wave * 16 + r + (hi << 3);
        #pragma unroll
        for (int t = 0; t < 4; ++t) {
            const int col = colbase + t * 16 + nloc;
            out[(size_t)row * D_ + col] = acc[t][r] + bo[col];
        }
    }
}

// ---------------------------------------------------------------------------
extern "C" void kernel_launch(void* const* d_in, const int* in_sizes, int n_in,
                              void* d_out, int out_size, void* d_ws, size_t ws_size,
                              hipStream_t stream) {
    const float* x            = (const float*)d_in[0];
    const int*   padding_mask = (const int*)d_in[1];
    const int*   future_mask  = (const int*)d_in[2];
    const float* Wq           = (const float*)d_in[3];
    const float* Wk           = (const float*)d_in[4];
    const float* Wv           = (const float*)d_in[5];
    const float* Wo           = (const float*)d_in[6];
    const float* bo           = (const float*)d_in[7];
    float*       out          = (float*)d_out;

    // Workspace layout (bf16): Q,K,V (8 MiB each), concat (8 MiB), Wo (2 MiB)
    const size_t qkvElems = (size_t)B_ * H_ * L_ * HD_;   // 4M
    __bf16* Qw  = (__bf16*)d_ws;
    __bf16* Kw  = Qw + qkvElems;
    __bf16* Vw  = Kw + qkvElems;
    __bf16* Cw  = Vw + qkvElems;
    __bf16* Wob = Cw + (size_t)B_ * L_ * D_;

    cvt_wo_kernel<<<(D_ * D_ + 255) / 256, 256, 0, stream>>>(Wo, Wob);

    dim3 g1(L_ / 128, B_ * H_);
    qkv_proj_kernel<<<g1, 256, 0, stream>>>(x, Wq, Wk, Wv, Qw, Kw, Vw);

    flash_attn_kernel<<<B_ * H_ * (L_ / 64), 128, 0, stream>>>(
        Qw, Kw, Vw, padding_mask, future_mask, Cw);

    dim3 g3((B_ * L_) / 64, D_ / 64);
    out_proj_kernel<<<g3, 128, 0, stream>>>(Cw, Wob, bo, out);
}
